// SpikingDenseLayer_1571958030366
// MI455X (gfx1250) — matrix operand order repaired and checked by reading, not compiled, not run on hardware
//
#include <hip/hip_runtime.h>
#include <hip/hip_bf16.h>

// ---------------------------------------------------------------------------
// SpikingDenseLayer forward for MI455X (gfx1250, wave32, WMMA bf16/f32-acc).
//
//  h   = x @ w                      (bf16 WMMA, f32 accum, async-LDS B staging)
//  d   = w^T @ w                    (same GEMM kernel)
//  recurrence over T:  rst = spk@d; mem=(mem-rst)*beta + h_t*(1-beta);
//                      spk = (mem*inv_norm - b > 0)
//  loss = 0.5 * mean(spk)  (exact integer spike count)
// ---------------------------------------------------------------------------

typedef __bf16 bf16_t;
typedef __attribute__((address_space(3))) __bf16 lds_bf16_t;
typedef __attribute__((ext_vector_type(16))) __bf16 v16bf;
typedef __attribute__((ext_vector_type(8)))  __bf16 v8bf;
typedef __attribute__((ext_vector_type(8)))  float  v8f;

// Problem shape (fixed by the reference)
static constexpr int Bsz = 128;
static constexpr int T   = 128;
static constexpr int I   = 1024;
static constexpr int O   = 1024;
static constexpr int M   = Bsz * T;      // 16384 rows of the h-GEMM
static constexpr int KT_I = I / 32;      // 32 K-tiles (K dim = I)
static constexpr int NT_O = O / 16;      // 64 N-tiles

__device__ inline v8f zero8() {
  v8f z;
#pragma unroll
  for (int i = 0; i < 8; ++i) z[i] = 0.0f;
  return z;
}

__device__ inline void wait_asynccnt0() {
#if __has_builtin(__builtin_amdgcn_s_wait_asynccnt)
  __builtin_amdgcn_s_wait_asynccnt(0);
#else
  asm volatile("s_wait_asynccnt 0x0" ::: "memory");
#endif
}

// --- A-fragment (16x32 bf16, MxK) loaded straight from row-major memory.
// Code's own k order (B packed to match): lane l holds row M=l%16; half=l/16 selects K-phase.
// Lane's 16 values = A[row, half*16 .. +15]:
// one contiguous 32-byte run -> two global_load_b128.
__device__ inline v16bf load_A_frag(const bf16_t* base, int lda, int mt, int kt) {
  const int lane = threadIdx.x & 31;
  const int row  = mt * 16 + (lane & 15);
  const int half = lane >> 4;
  const bf16_t* p = base + (size_t)row * lda + kt * 32 + half * 16;
  v8bf lo = *reinterpret_cast<const v8bf*>(p);
  v8bf hi = *reinterpret_cast<const v8bf*>(p + 8);
  v16bf a;
#pragma unroll
  for (int i = 0; i < 8; ++i) { a[i] = lo[i]; a[8 + i] = hi[i]; }
  return a;
}

// --- B-fragment (32x16 bf16, KxN): lane owns 16 contiguous bf16 (32 bytes).
__device__ inline v16bf load_B_frag(const bf16_t* pack, int NT, int kt, int nt) {
  const int lane = threadIdx.x & 31;
  const bf16_t* p = pack + ((((size_t)kt * NT + nt) * 32) + lane) * 16;
  v8bf lo = *reinterpret_cast<const v8bf*>(p);
  v8bf hi = *reinterpret_cast<const v8bf*>(p + 8);
  v16bf b;
#pragma unroll
  for (int i = 0; i < 8; ++i) { b[i] = lo[i]; b[8 + i] = hi[i]; }
  return b;
}

// --- B-fragment gathered from an LDS-staged fragment (ds_load_b128 x2).
__device__ inline v16bf load_B_frag_lds(const bf16_t* fragBase) {
  const int lane = threadIdx.x & 31;
  const bf16_t* p = fragBase + lane * 16;
  v8bf lo = *reinterpret_cast<const v8bf*>(p);
  v8bf hi = *reinterpret_cast<const v8bf*>(p + 8);
  v16bf b;
#pragma unroll
  for (int i = 0; i < 8; ++i) { b[i] = lo[i]; b[8 + i] = hi[i]; }
  return b;
}

// --- C/D 16x16 f32 tile store (lanes 0-15: M=v, lanes 16-31: M=8+v; N=lane%16)
__device__ inline void store_C(float* C, int ldc, int mt, int nt, v8f c) {
  const int lane = threadIdx.x & 31;
  const int n    = lane & 15;
  const int half = lane >> 4;
#pragma unroll
  for (int v = 0; v < 8; ++v)
    C[(size_t)(mt * 16 + half * 8 + v) * ldc + nt * 16 + n] = c[v];
}

// ---------------------------------------------------------------------------
// Utility kernels
// ---------------------------------------------------------------------------

__global__ void zero_kernel(unsigned int* p, size_t nwords) {
  size_t i = (size_t)blockIdx.x * blockDim.x + threadIdx.x;
  size_t stride = (size_t)gridDim.x * blockDim.x;
  for (; i < nwords; i += stride) p[i] = 0u;
}

__global__ void cvt_f32_bf16(const float* in, bf16_t* out, size_t n) {
  size_t i = (size_t)blockIdx.x * blockDim.x + threadIdx.x;
  if (i < n) out[i] = (bf16_t)in[i];
}

// Transpose w [I,O] f32 -> wT bf16 row-major [O,I] (A-operand of d = w^T w)
__global__ void transpose_w(const float* w, bf16_t* wT) {
  size_t gid = (size_t)blockIdx.x * blockDim.x + threadIdx.x;
  if (gid >= (size_t)I * O) return;
  int m = (int)(gid >> 10);        // O index (row of wT)
  int k = (int)(gid & 1023);       // I index
  wT[(size_t)m * I + k] = (bf16_t)w[(size_t)k * O + m];
}

// Pack f32 row-major [K,N] into B-fragment-major bf16 (contiguous-k order, same as load_A_frag):
// fragment (kt,nt), lane l (N=l%16, halfK=l/16) holds the contiguous
// K-column segment  in[kt*32 + halfK*16 + e, nt*16 + n], e = 0..15.
__global__ void pack_B_f32(const float* in, bf16_t* out, int K, int N) {
  int gid = blockIdx.x * blockDim.x + threadIdx.x;
  const int NT = N / 16;
  const int total = (K / 32) * NT * 32;
  if (gid >= total) return;
  const int lane = gid & 31;
  const int frag = gid >> 5;
  const int nt = frag % NT;
  const int kt = frag / NT;
  const int n = lane & 15;
  const int halfK = lane >> 4;
  v8bf lo, hi;
#pragma unroll
  for (int e = 0; e < 8; ++e)
    lo[e] = (bf16_t)in[(size_t)(kt * 32 + halfK * 16 + e) * N + nt * 16 + n];
#pragma unroll
  for (int e = 0; e < 8; ++e)
    hi[e] = (bf16_t)in[(size_t)(kt * 32 + halfK * 16 + 8 + e) * N + nt * 16 + n];
  bf16_t* o = out + (size_t)gid * 16;
  *reinterpret_cast<v8bf*>(o)     = lo;
  *reinterpret_cast<v8bf*>(o + 8) = hi;
}

// inv_norm[o] = 1 / (sum_k w[k,o]^2 + eps); one wave per output column.
__global__ void norm_kernel(const float* w, float* inv_norm) {
  const int lane = threadIdx.x & 31;
  const int o = blockIdx.x * 8 + (threadIdx.x >> 5);
  float s = 0.0f;
  for (int k = lane; k < I; k += 32) {
    float v = w[(size_t)k * O + o];
    s += v * v;
  }
#pragma unroll
  for (int off = 16; off > 0; off >>= 1) s += __shfl_down(s, off, 32);
  if (lane == 0) inv_norm[o] = 1.0f / (s + 1e-8f);
}

// ---------------------------------------------------------------------------
// bf16 WMMA GEMM with async global->LDS B staging.
// Block = 8 waves sharing one 64-column B strip; wave w owns rows of tile
// mtBase+w. Per K-tile the 4 KB contiguous B chunk (4 fragments) is DMA'd to
// LDS with one global_load_async_to_lds_b128 per thread, double-buffered so
// the next chunk's async copy overlaps the current chunk's 4 WMMAs.
// Requires Mtiles % 8 == 0.
// ---------------------------------------------------------------------------
__global__ void gemm_bf16_lds(const bf16_t* __restrict__ A, int lda,
                              const bf16_t* __restrict__ Bpack,
                              float* __restrict__ C, int ldc,
                              int Mtiles, int NT, int Ktiles) {
  __shared__ bf16_t ldsB[2 * 2048];          // 2 x 4KB double buffer
  const int NTg  = NT / 4;
  const int tid  = threadIdx.x;
  const int wave = tid >> 5;
  const int mt   = (blockIdx.x / NTg) * 8 + wave;
  const int nt0  = (blockIdx.x % NTg) * 4;
  (void)Mtiles;

  // One async b128 per thread moves the whole 4 KB chunk for K-tile kt.
  auto issue_chunk = [&](int kt, int buf) {
    const bf16_t* src = Bpack + (((size_t)kt * NT + nt0) * 32) * 16 + tid * 8;
    unsigned dst =
        (unsigned)(uintptr_t)(lds_bf16_t*)&ldsB[buf * 2048 + tid * 8];
    asm volatile("global_load_async_to_lds_b128 %0, %1, off"
                 :: "v"(dst), "v"(src) : "memory");
  };

  issue_chunk(0, 0);

  v8f c0 = zero8(), c1 = zero8(), c2 = zero8(), c3 = zero8();
  for (int kt = 0; kt < Ktiles; ++kt) {
    wait_asynccnt0();          // this wave's chunk for buf[kt&1] has landed
    __syncthreads();           // everyone's chunk landed; prev buf fully read
    if (kt + 1 < Ktiles) issue_chunk(kt + 1, (kt + 1) & 1);

    v16bf a = load_A_frag(A, lda, mt, kt);
    const bf16_t* bb = &ldsB[(kt & 1) * 2048];
    v16bf b0 = load_B_frag_lds(bb + 0 * 512);
    v16bf b1 = load_B_frag_lds(bb + 1 * 512);
    v16bf b2 = load_B_frag_lds(bb + 2 * 512);
    v16bf b3 = load_B_frag_lds(bb + 3 * 512);
    c0 = __builtin_amdgcn_wmma_f32_16x16x32_bf16(false, a, false, b0, (short)0, c0, false, false);
    c1 = __builtin_amdgcn_wmma_f32_16x16x32_bf16(false, a, false, b1, (short)0, c1, false, false);
    c2 = __builtin_amdgcn_wmma_f32_16x16x32_bf16(false, a, false, b2, (short)0, c2, false, false);
    c3 = __builtin_amdgcn_wmma_f32_16x16x32_bf16(false, a, false, b3, (short)0, c3, false, false);
  }
  store_C(C, ldc, mt, nt0 + 0, c0);
  store_C(C, ldc, mt, nt0 + 1, c1);
  store_C(C, ldc, mt, nt0 + 2, c2);
  store_C(C, ldc, mt, nt0 + 3, c3);
}

// ---------------------------------------------------------------------------
// One recurrence step t:  rst = spk_prev @ d  (WMMA), then elementwise update.
// 128 waves total: 8 batch-tiles x 16 n-groups; each wave owns 16x64 of [B,O].
// Bd (2 MB) is re-read every step by only 128 waves -> stays hot in L2;
// direct global loads are the right path here.
// ---------------------------------------------------------------------------
__global__ void step_kernel(const bf16_t* __restrict__ spk_prev,   // [B,O] bf16
                            const bf16_t* __restrict__ Bd,         // packed d
                            const float*  __restrict__ h,          // [M,O]
                            float*        __restrict__ mem,        // [B,O]
                            const float*  __restrict__ inv_norm,   // [O]
                            const float*  __restrict__ bvec,       // [O]
                            const float*  __restrict__ beta_p,     // [1]
                            float*        __restrict__ spk_rec,    // d_out [B,T,O]
                            bf16_t*       __restrict__ spk_next,   // [B,O] bf16
                            unsigned int* __restrict__ counter,
                            int t) {
  const int wave = (int)((blockIdx.x * blockDim.x + threadIdx.x) >> 5);
  if (wave >= 8 * 16) return;
  const int bt  = wave / 16;          // batch tile (rows bt*16..)
  const int nt0 = (wave % 16) * 4;    // first of 4 N-tiles

  v8f acc[4] = {zero8(), zero8(), zero8(), zero8()};
  for (int kt = 0; kt < KT_I; ++kt) {
    v16bf a = load_A_frag(spk_prev, O, bt, kt);
#pragma unroll
    for (int j = 0; j < 4; ++j) {
      v16bf b = load_B_frag(Bd, NT_O, kt, nt0 + j);
      acc[j] = __builtin_amdgcn_wmma_f32_16x16x32_bf16(false, a, false, b, (short)0, acc[j], false, false);
    }
  }

  const float beta = beta_p[0];
  const float omb  = 1.0f - beta;
  const int lane = threadIdx.x & 31;
  const int n    = lane & 15;
  const int half = lane >> 4;
  unsigned int cnt = 0;

#pragma unroll
  for (int j = 0; j < 4; ++j) {
    const int o = (nt0 + j) * 16 + n;
    const float invn = inv_norm[o];
    const float bb   = bvec[o];
#pragma unroll
    for (int v = 0; v < 8; ++v) {
      const int bi = bt * 16 + half * 8 + v;
      const size_t mi = (size_t)bi * O + o;
      const size_t hi = ((size_t)bi * T + t) * O + o;
      float m2 = (mem[mi] - acc[j][v]) * beta + h[hi] * omb;
      float mthr = m2 * invn - bb;
      float s = (mthr > 0.0f) ? 1.0f : 0.0f;
      mem[mi]      = m2;
      spk_rec[hi]  = s;
      spk_next[mi] = (bf16_t)s;
      cnt += (mthr > 0.0f) ? 1u : 0u;
    }
  }
  atomicAdd(counter, cnt);
}

__global__ void loss_kernel(const unsigned int* counter, float* out, size_t idx) {
  if (threadIdx.x == 0 && blockIdx.x == 0)
    out[idx] = 0.5f * (float)(*counter) / (float)((size_t)M * O);
}

// ---------------------------------------------------------------------------
// Host launcher
// ---------------------------------------------------------------------------
extern "C" void kernel_launch(void* const* d_in, const int* in_sizes, int n_in,
                              void* d_out, int out_size, void* d_ws, size_t ws_size,
                              hipStream_t stream) {
  (void)in_sizes; (void)n_in; (void)ws_size;
  const float* x    = (const float*)d_in[0];   // [B,T,I]
  const float* w    = (const float*)d_in[1];   // [I,O]
  const float* beta = (const float*)d_in[2];   // [1]
  const float* bv   = (const float*)d_in[3];   // [O]
  float* out = (float*)d_out;                  // spk_rec flat + loss

  // Workspace layout
  char* ws = (char*)d_ws;
  size_t off = 0;
  bf16_t* x_bf   = (bf16_t*)(ws + off); off += (size_t)M * I * 2;      // 32 MB
  float*  h      = (float*) (ws + off); off += (size_t)M * O * 4;      // 64 MB
  bf16_t* Bw     = (bf16_t*)(ws + off); off += (size_t)I * O * 2;      //  2 MB
  bf16_t* wT     = (bf16_t*)(ws + off); off += (size_t)O * I * 2;      //  2 MB
  float*  d_f32  = (float*) (ws + off); off += (size_t)O * O * 4;      //  4 MB
  bf16_t* Bd     = (bf16_t*)(ws + off); off += (size_t)O * O * 2;      //  2 MB
  float*  mem    = (float*) (ws + off); off += (size_t)Bsz * O * 4;    // 512 KB
  bf16_t* spk0   = (bf16_t*)(ws + off); off += (size_t)Bsz * O * 2;    // 256 KB
  bf16_t* spk1   = (bf16_t*)(ws + off); off += (size_t)Bsz * O * 2;    // 256 KB
  float*  invn   = (float*) (ws + off); off += (size_t)O * 4;          //   4 KB
  unsigned int* counter = (unsigned int*)(ws + off); off += 4;

  // 1) zero mem, spk ping-pong buffers, inv_norm slot, counter (contiguous)
  {
    size_t bytes = (size_t)Bsz * O * 4 + (size_t)Bsz * O * 2 * 2 + (size_t)O * 4 + 4;
    size_t nwords = bytes / 4;
    int blk = 256, grd = (int)((nwords + blk - 1) / blk);
    zero_kernel<<<grd, blk, 0, stream>>>((unsigned int*)mem, nwords);
  }
  // 2) x -> bf16 row-major [M, I]
  {
    size_t n = (size_t)M * I;
    cvt_f32_bf16<<<(int)((n + 255) / 256), 256, 0, stream>>>(x, x_bf, n);
  }
  // 3) pack w into B-fragment layout (serves both h-GEMM and d-GEMM)
  pack_B_f32<<<(KT_I * NT_O * 32) / 256, 256, 0, stream>>>(w, Bw, I, O);
  // 4) wT bf16 (A-operand of d = w^T w)
  transpose_w<<<(I * O) / 256, 256, 0, stream>>>(w, wT);
  // 5) inv_norm
  norm_kernel<<<O / 8, 256, 0, stream>>>(w, invn);
  // 6) h = x @ w   (Mtiles=1024, NT=64, Ktiles=32): 128 mt-blocks x 16 nt-groups
  gemm_bf16_lds<<<(M / 16 / 8) * (NT_O / 4), 256, 0, stream>>>(
      x_bf, I, Bw, h, O, M / 16, NT_O, KT_I);
  // 7) d = w^T @ w  (Mtiles=64): 8 mt-blocks x 16 nt-groups
  gemm_bf16_lds<<<(O / 16 / 8) * (NT_O / 4), 256, 0, stream>>>(
      wT, I, Bw, d_f32, O, O / 16, NT_O, KT_I);
  // 8) pack d into B-fragment layout for the recurrence
  pack_B_f32<<<(KT_I * NT_O * 32) / 256, 256, 0, stream>>>(d_f32, Bd, O, O);
  // 9) T sequential steps, ping-pong spike buffers (t=0 reads zeroed spk0)
  for (int t = 0; t < T; ++t) {
    bf16_t* sp = (t & 1) ? spk1 : spk0;
    bf16_t* sn = (t & 1) ? spk0 : spk1;
    step_kernel<<<16, 256, 0, stream>>>(sp, Bd, h, mem, invn, bv, beta,
                                        out, sn, counter, t);
  }
  // 10) loss = 0.5 * count / (B*T*O) appended after spk_rec
  loss_kernel<<<1, 64, 0, stream>>>(counter, out, (size_t)out_size - 1);
}
